// TabularDiffFlow_86028194939123
// MI455X (gfx1250) — matrix operand order repaired and checked by reading, not compiled, not run on hardware
//
#include <hip/hip_runtime.h>
#include <hip/hip_bf16.h>

// ---------------------------------------------------------------------------
// MI455X (gfx1250) fused attention pipeline.
// Heavy GEMMs + attention matmuls use v_wmma_f32_16x16x32_bf16 (wave32 WMMA).
// scores [B,H,N,N] (134 MB) never touch HBM: held per-32-row-block in LDS.
// GEMM wave tile 2x4 (32x64 output / wave): 6 fragment loads per 8 WMMAs.
// Workspace requirement: ~86 MB (bf16 activations/weights + pv gate scalars).
// ---------------------------------------------------------------------------

typedef __attribute__((ext_vector_type(16))) __bf16 bf16x16;
typedef __attribute__((ext_vector_type(8)))  __bf16 bf16x8;
typedef __attribute__((ext_vector_type(8)))  float  f32x8;

#define BQ   64
#define NSEQ 256
#define DMOD 512
#define NH   8
#define HD   64

__device__ __forceinline__ unsigned short f2bf(float f) {
  unsigned u = __float_as_uint(f);
  u += 0x7FFFu + ((u >> 16) & 1u);          // round-to-nearest-even
  return (unsigned short)(u >> 16);
}

// orderable-uint mapping for float comparison (handles negatives)
__device__ __forceinline__ unsigned ford(float f) {
  unsigned u = __float_as_uint(f);
  return (u & 0x80000000u) ? ~u : (u | 0x80000000u);
}

// A-operand fragment, 16x32 bf16 (contiguous k = 16*(lane>>4) + s order, same as ldB).
// p = row_ptr + kbase + (lane>>4)*16 ; 16 contiguous K halfs.
__device__ __forceinline__ bf16x16 ldA(const unsigned short* p) {
  bf16x8 lo = *(const bf16x8*)(p);
  bf16x8 hi = *(const bf16x8*)(p + 8);
  return __builtin_shufflevector(lo, hi, 0,1,2,3,4,5,6,7,8,9,10,11,12,13,14,15);
}
// B-operand fragment, 32x16 bf16: lane = column, 16 contiguous K halfs.
__device__ __forceinline__ bf16x16 ldB(const unsigned short* p) {
  return *(const bf16x16*)p;
}

__device__ __forceinline__ f32x8 wmma_bf16(bf16x16 a, bf16x16 b, f32x8 c) {
  return __builtin_amdgcn_wmma_f32_16x16x32_bf16(false, a, false, b,
                                                 (short)0, c, false, false);
}

// ---------------------------------------------------------------------------
// prep: f32 -> bf16 elementwise
__global__ __launch_bounds__(256)
void cvt_bf16_kernel(const float* __restrict__ in, unsigned short* __restrict__ out, int n) {
  int i = blockIdx.x * 256 + threadIdx.x;
  if (i < n) out[i] = f2bf(in[i]);
}

// prep: W[512,512] f32 row-major -> WT[j][d] bf16 (column of W contiguous)
__global__ __launch_bounds__(256)
void transpose_w_kernel(const float* __restrict__ W, unsigned short* __restrict__ WT) {
  int idx = blockIdx.x * 256 + threadIdx.x;       // 512*512 threads
  int d = idx >> 9, j = idx & 511;
  WT[(size_t)j * DMOD + d] = f2bf(W[(size_t)d * DMOD + j]);
}

// ---------------------------------------------------------------------------
// pv gate: pvc[b] = 0.5 * sigmoid( silu(combined @ W1 + b1) @ W2 + b2 )
__global__ __launch_bounds__(256)
void pv_kernel(const float* __restrict__ x, const float* __restrict__ W1,
               const float* __restrict__ b1, const float* __restrict__ W2,
               const float* __restrict__ b2, float* __restrict__ pvc) {
  __shared__ float comb[2 * DMOD];
  __shared__ float acc_logit;
  int b = blockIdx.x, tid = threadIdx.x;
  const float* xb = x + (size_t)b * NSEQ * DMOD;
  float p0 = 0.f, p1 = 0.f, v0 = 0.f, v1 = 0.f;
  for (int n = 0; n < 128; ++n)  { p0 += xb[n*DMOD + tid]; p1 += xb[n*DMOD + tid + 256]; }
  for (int n = 128; n < 256; ++n){ v0 += xb[n*DMOD + tid]; v1 += xb[n*DMOD + tid + 256]; }
  const float inv128 = 1.f / 128.f;
  comb[tid]        = p0 * inv128;  comb[tid + 256] = p1 * inv128;
  comb[512 + tid]  = v0 * inv128;  comb[768 + tid] = v1 * inv128;
  if (tid == 0) acc_logit = 0.f;
  __syncthreads();
  float part = 0.f;
  for (int rep = 0; rep < 2; ++rep) {
    int j = tid + rep * 256;
    float s = b1[j];
    for (int t = 0; t < 1024; ++t) s += comb[t] * W1[(size_t)t * DMOD + j];
    float sig = 1.f / (1.f + __expf(-s));       // silu = s * sigmoid(s)
    part += (s * sig) * W2[j];
  }
  atomicAdd(&acc_logit, part);
  __syncthreads();
  if (tid == 0) {
    float logit = acc_logit + b2[0];
    pvc[b] = 0.5f / (1.f + __expf(-logit));     // 0.5 * sigmoid folded in
  }
}

// ---------------------------------------------------------------------------
// GEMM: Y[M,N] = A[M,K](bf16) * WT[N,K]^T(bf16) + bias, M=16384,N=K=512.
// Block tile 128x128 (8 m-tiles x 8 n-tiles); wave w owns 2 m-tiles x 4
// n-tiles: mt0=(w&3)*2, nt0=(w>>2)*4. 6 fragment loads feed 8 WMMAs / k-step.
// mode 0: bf16 row-major out; mode 1: bf16 transposed out [B, D, NSEQ] (for v);
// mode 2: f32 row-major out (final projection into d_out).
__global__ __launch_bounds__(256)
void gemm_wmma_kernel(const unsigned short* __restrict__ A,
                      const unsigned short* __restrict__ WT,
                      const float* __restrict__ bias,
                      void* __restrict__ out, int M, int N, int K, int mode) {
  int w = threadIdx.x >> 5;
  int lane = threadIdx.x & 31;
  int hf = lane >> 4, lj = lane & 15;
  int mblk = blockIdx.x * 128;
  int nblk = blockIdx.y * 128;
  int mt0 = (w & 3) * 2;
  int nt0 = (w >> 2) * 4;

  f32x8 acc[2][4] = {};
  const unsigned short* arow0 = A + (size_t)(mblk + (mt0 + 0)*16 + lj) * K;
  const unsigned short* arow1 = A + (size_t)(mblk + (mt0 + 1)*16 + lj) * K;
  const unsigned short* brow0 = WT + (size_t)(nblk + (nt0 + 0)*16 + lj) * K;
  const unsigned short* brow1 = WT + (size_t)(nblk + (nt0 + 1)*16 + lj) * K;
  const unsigned short* brow2 = WT + (size_t)(nblk + (nt0 + 2)*16 + lj) * K;
  const unsigned short* brow3 = WT + (size_t)(nblk + (nt0 + 3)*16 + lj) * K;

  for (int kk = 0; kk < K; kk += 32) {
    bf16x16 a0 = ldA(arow0 + kk + hf*16);
    bf16x16 a1 = ldA(arow1 + kk + hf*16);
    bf16x16 b0 = ldB(brow0 + kk + hf*16);
    bf16x16 b1 = ldB(brow1 + kk + hf*16);
    bf16x16 b2 = ldB(brow2 + kk + hf*16);
    bf16x16 b3 = ldB(brow3 + kk + hf*16);
    acc[0][0] = wmma_bf16(a0, b0, acc[0][0]);
    acc[1][0] = wmma_bf16(a1, b0, acc[1][0]);
    acc[0][1] = wmma_bf16(a0, b1, acc[0][1]);
    acc[1][1] = wmma_bf16(a1, b1, acc[1][1]);
    acc[0][2] = wmma_bf16(a0, b2, acc[0][2]);
    acc[1][2] = wmma_bf16(a1, b2, acc[1][2]);
    acc[0][3] = wmma_bf16(a0, b3, acc[0][3]);
    acc[1][3] = wmma_bf16(a1, b3, acc[1][3]);
  }
  #pragma unroll
  for (int mi = 0; mi < 2; ++mi) {
    #pragma unroll
    for (int t = 0; t < 4; ++t) {
      int gn = nblk + (nt0 + t)*16 + lj;
      float bv = bias[gn];
      #pragma unroll
      for (int r = 0; r < 8; ++r) {
        int gm = mblk + (mt0 + mi)*16 + hf*8 + r;  // lanes 16-31 hold M=r+8
        float v = acc[mi][t][r] + bv;
        if (mode == 0) {
          ((unsigned short*)out)[(size_t)gm * N + gn] = f2bf(v);
        } else if (mode == 1) {                    // vT[b][d][n]
          int bb = gm >> 8, n = gm & 255;
          ((unsigned short*)out)[((size_t)bb * N + gn) * NSEQ + n] = f2bf(v);
        } else {
          ((float*)out)[(size_t)gm * N + gn] = v;
        }
      }
    }
  }
}

// ---------------------------------------------------------------------------
// Fused attention: one block per (b,h). 8 row-blocks of 32 rows.
// S (f32, LDS, stride 257) -> prior/bias -> exact top-128 threshold via
// 32-step bitwise binary search (8 lanes/row, shfl_xor combine) -> softmax ->
// P (bf16, LDS) @ vT via WMMA -> bf16 out [B*N, D].
__global__ __launch_bounds__(256)
void attn_kernel(const unsigned short* __restrict__ q,
                 const unsigned short* __restrict__ k,
                 const unsigned short* __restrict__ vT,
                 const float* __restrict__ corr_prior,
                 const float* __restrict__ feat_imp,
                 const float* __restrict__ pvc,
                 unsigned short* __restrict__ out) {
  __shared__ __attribute__((aligned(32))) float          Ssc[32 * 257];
  __shared__ __attribute__((aligned(32))) unsigned short Pm[32 * 264];

  int bh = blockIdx.x;
  int b = bh >> 3, h = bh & 7;
  int tid = threadIdx.x;
  int w = tid >> 5, lane = tid & 31, hf = lane >> 4, lj = lane & 15;

  const unsigned short* qb = q  + (size_t)b * NSEQ * DMOD + h * HD;
  const unsigned short* kb = k  + (size_t)b * NSEQ * DMOD + h * HD;
  const unsigned short* vb = vT + (size_t)b * DMOD * NSEQ + (size_t)h * HD * NSEQ;
  float pv = pvc[b];

  for (int rb = 0; rb < 8; ++rb) {
    int row0 = rb * 32;

    // ---- phase 1: S = (q kT) * 0.125 into LDS --------------------------
    {
      int mt = w & 1;
      int nt0 = (w >> 1) * 4;
      f32x8 acc[4] = {};
      const unsigned short* arow = qb + (size_t)(row0 + mt*16 + lj) * DMOD;
      for (int kk = 0; kk < HD; kk += 32) {
        bf16x16 a = ldA(arow + kk + hf*16);
        #pragma unroll
        for (int t = 0; t < 4; ++t) {
          const unsigned short* bp = kb + (size_t)((nt0 + t)*16 + lj) * DMOD + kk + hf*16;
          acc[t] = wmma_bf16(a, ldB(bp), acc[t]);
        }
      }
      #pragma unroll
      for (int t = 0; t < 4; ++t)
        #pragma unroll
        for (int r = 0; r < 8; ++r)
          Ssc[(mt*16 + hf*8 + r) * 257 + (nt0 + t)*16 + lj] = acc[t][r] * 0.125f;
    }
    __syncthreads();

    // ---- phase 2: priors + exact kth-largest (k=128) + softmax ---------
    int row = tid >> 3, sub = tid & 7;        // 8 lanes per row, same wave octet
    int gi = row0 + row;
    float impi = feat_imp[gi];
    const float* cp = corr_prior + (size_t)gi * NSEQ + sub * 32;
    float s[32];
    unsigned u[32];
    #pragma unroll
    for (int c = 0; c < 32; ++c) {
      int j = sub * 32 + c;
      float cross = ((gi < 128) != (j < 128)) ? pv : 0.f;
      s[c] = Ssc[row * 257 + j] + cp[c] * impi * feat_imp[j] + cross;
      u[c] = ford(s[c]);
    }
    // bitwise binary search on orderable bits: cur -> exact 128th largest
    unsigned cur = 0u;
    for (int bit = 31; bit >= 0; --bit) {
      unsigned cand = cur | (1u << bit);
      int cnt = 0;
      #pragma unroll
      for (int c = 0; c < 32; ++c) cnt += (u[c] >= cand) ? 1 : 0;
      cnt += __shfl_xor(cnt, 1);
      cnt += __shfl_xor(cnt, 2);
      cnt += __shfl_xor(cnt, 4);
      if (cnt >= 128) cur = cand;
    }
    // row max via the same orderable ints (order-preserving)
    unsigned um = 0u;
    #pragma unroll
    for (int c = 0; c < 32; ++c) um = (u[c] > um) ? u[c] : um;
    um = max(um, (unsigned)__shfl_xor((int)um, 1));
    um = max(um, (unsigned)__shfl_xor((int)um, 2));
    um = max(um, (unsigned)__shfl_xor((int)um, 4));
    float m = __uint_as_float((um & 0x80000000u) ? (um & 0x7FFFFFFFu) : ~um);
    float p[32], psum = 0.f;
    #pragma unroll
    for (int c = 0; c < 32; ++c) {
      float e = (u[c] >= cur) ? __expf(s[c] - m) : 0.f;
      p[c] = e; psum += e;
    }
    psum += __shfl_xor(psum, 1);
    psum += __shfl_xor(psum, 2);
    psum += __shfl_xor(psum, 4);
    float inv = 1.f / psum;
    #pragma unroll
    for (int c = 0; c < 32; ++c)
      Pm[row * 264 + sub * 32 + c] = f2bf(p[c] * inv);
    __syncthreads();

    // ---- phase 3: O = P @ v via WMMA (2 m-tiles x 4 n-tiles, 1/wave) ---
    {
      int mt = w & 1, nt = w >> 1;
      f32x8 acc = {};
      const unsigned short* arow = Pm + (size_t)(mt*16 + lj) * 264;
      const unsigned short* bcol = vb + (size_t)(nt*16 + lj) * NSEQ;
      for (int kk = 0; kk < NSEQ; kk += 32)
        acc = wmma_bf16(ldA(arow + kk + hf*16), ldB(bcol + kk + hf*16), acc);
      #pragma unroll
      for (int r = 0; r < 8; ++r) {
        int gm = b * NSEQ + row0 + mt*16 + hf*8 + r;
        int gn = h * HD + nt*16 + lj;
        out[(size_t)gm * DMOD + gn] = f2bf(acc[r]);
      }
    }
    __syncthreads();
  }
}

// ---------------------------------------------------------------------------
extern "C" void kernel_launch(void* const* d_in, const int* in_sizes, int n_in,
                              void* d_out, int out_size, void* d_ws, size_t ws_size,
                              hipStream_t stream) {
  (void)in_sizes; (void)n_in; (void)out_size; (void)ws_size;
  const float* x    = (const float*)d_in[0];
  const float* Wq   = (const float*)d_in[1];
  const float* bq   = (const float*)d_in[2];
  const float* Wk   = (const float*)d_in[3];
  const float* bk   = (const float*)d_in[4];
  const float* Wv   = (const float*)d_in[5];
  const float* bv   = (const float*)d_in[6];
  const float* Wo   = (const float*)d_in[7];
  const float* bo   = (const float*)d_in[8];
  const float* corr = (const float*)d_in[9];
  const float* fimp = (const float*)d_in[10];
  const float* W1   = (const float*)d_in[11];
  const float* b1   = (const float*)d_in[12];
  const float* W2   = (const float*)d_in[13];
  const float* b2   = (const float*)d_in[14];

  constexpr size_t NX = (size_t)BQ * NSEQ * DMOD;   // 8,388,608
  constexpr size_t NW = (size_t)DMOD * DMOD;        //   262,144

  unsigned short* ws  = (unsigned short*)d_ws;
  size_t off = 0;
  unsigned short* xb   = ws + off; off += NX;
  unsigned short* WqT  = ws + off; off += NW;
  unsigned short* WkT  = ws + off; off += NW;
  unsigned short* WvT  = ws + off; off += NW;
  unsigned short* WoT  = ws + off; off += NW;
  unsigned short* qbuf = ws + off; off += NX;
  unsigned short* kbuf = ws + off; off += NX;
  unsigned short* vTb  = ws + off; off += NX;
  unsigned short* aout = ws + off; off += NX;
  float* pvc = (float*)(ws + off);                  // 64 floats

  // prep
  cvt_bf16_kernel<<<(int)(NX / 256), 256, 0, stream>>>(x, xb, (int)NX);
  transpose_w_kernel<<<(int)(NW / 256), 256, 0, stream>>>(Wq, WqT);
  transpose_w_kernel<<<(int)(NW / 256), 256, 0, stream>>>(Wk, WkT);
  transpose_w_kernel<<<(int)(NW / 256), 256, 0, stream>>>(Wv, WvT);
  transpose_w_kernel<<<(int)(NW / 256), 256, 0, stream>>>(Wo, WoT);
  pv_kernel<<<BQ, 256, 0, stream>>>(x, W1, b1, W2, b2, pvc);

  // QKV projections (block tile 128x128)
  dim3 ggrid(BQ * NSEQ / 128, DMOD / 128);
  gemm_wmma_kernel<<<ggrid, 256, 0, stream>>>(xb, WqT, bq, qbuf, BQ*NSEQ, DMOD, DMOD, 0);
  gemm_wmma_kernel<<<ggrid, 256, 0, stream>>>(xb, WkT, bk, kbuf, BQ*NSEQ, DMOD, DMOD, 0);
  gemm_wmma_kernel<<<ggrid, 256, 0, stream>>>(xb, WvT, bv, vTb,  BQ*NSEQ, DMOD, DMOD, 1);

  // fused attention
  attn_kernel<<<BQ * NH, 256, 0, stream>>>(qbuf, kbuf, vTb, corr, fimp, pvc, aout);

  // output projection -> f32 d_out
  gemm_wmma_kernel<<<ggrid, 256, 0, stream>>>(aout, WoT, bo, d_out, BQ*NSEQ, DMOD, DMOD, 2);
}
